// ImplicitValueAttention_4793183502833
// MI455X (gfx1250) — matrix operand order repaired and checked by reading, not compiled, not run on hardware
//
#include <hip/hip_runtime.h>
#include <stdint.h>

// Problem shape (fixed by the reference)
#define BB 2
#define NN 4096
#define DD 1024
#define HH 8
#define HDIM 128
#define CK 64
#define NTOK (BB*NN)

typedef __attribute__((ext_vector_type(16))) __bf16 v16bf;
typedef __attribute__((ext_vector_type(8)))  float  v8f;

union BF16Frag { uint32_t u[8]; v16bf v; };

__device__ __forceinline__ uint32_t f2bf_bits(float a){
  uint32_t u = __float_as_uint(a);
  return (u + 0x7FFFu + ((u >> 16) & 1u)) >> 16;   // round-to-nearest-even
}
__device__ __forceinline__ uint32_t pack_bf16(float a, float b){
  return (f2bf_bits(b) << 16) | (f2bf_bits(a) & 0xFFFFu);
}
__device__ __forceinline__ float bf16_to_f32(uint16_t h){
  return __uint_as_float(((uint32_t)h) << 16);
}

__device__ __forceinline__ v8f wmma_bf16(v16bf a, v16bf b, v8f c){
  // D(f32 16x16) = A(bf16 16x32) x B(bf16 32x16) + C(f32)
  return __builtin_amdgcn_wmma_f32_16x16x32_bf16(false, a, false, b, (short)0, c, false, false);
}

__device__ __forceinline__ void wait_async0(){
#if defined(__has_builtin)
#if __has_builtin(__builtin_amdgcn_s_wait_asynccnt)
  __builtin_amdgcn_s_wait_asynccnt(0);
#else
  asm volatile("s_wait_asynccnt 0" ::: "memory");
#endif
#else
  asm volatile("s_wait_asynccnt 0" ::: "memory");
#endif
}

// Async DMA: copy 16 bytes per lane from global (bf16 data) into LDS.
// laddr = byte offset from the wave's LDS base (dynamic LDS starts at
// groupstaticsize()); gaddr = 64-bit global byte address. Tracked by ASYNCcnt.
__device__ __forceinline__ void async_ldst_b128(uint32_t laddr, uint64_t gaddr){
  asm volatile("global_load_async_to_lds_b128 %0, %1, off"
               :: "v"(laddr), "v"(gaddr) : "memory");
}

// ---- WMMA fragment loaders (16-bit A 16x32 layout per ISA 7.12.2) ----
// A fragment from A stored row-major (M x K), base -> A[m0][k0], stride ldt elements.
__device__ __forceinline__ v16bf load_afrag(const uint16_t* base, int ldt, int lane){
  int m = lane & 15, hf = lane >> 4;
  BF16Frag f;
#pragma unroll
  for (int v = 0; v < 8; ++v){
    int kb = (v < 4 ? 2*v : 16 + 2*(v - 4)) + 8*hf;
    f.u[v] = *(const uint32_t*)(base + m*ldt + kb);
  }
  return f.v;
}
// B fragment from B^T stored row-major (N x K): base -> B^T[n0][k0].
__device__ __forceinline__ v16bf load_bfrag_bt(const uint16_t* base, int ldt, int lane){
  int n = lane & 15, hf = lane >> 4;
  BF16Frag f;
#pragma unroll
  for (int v = 0; v < 8; ++v){
    int kb = (v < 4 ? 2*v : 16 + 2*(v - 4)) + 8*hf;
    f.u[v] = *(const uint32_t*)(base + n*ldt + kb);
  }
  return f.v;
}
// B fragment from B stored row-major (K x N): base -> B[k0][n0]. (strided 16-bit reads)
__device__ __forceinline__ v16bf load_bfrag_b(const uint16_t* base, int ldt, int lane){
  int n = lane & 15, hf = lane >> 4;
  BF16Frag f;
#pragma unroll
  for (int v = 0; v < 8; ++v){
    int kk = (v < 4 ? 2*v : 16 + 2*(v - 4)) + 8*hf;
    uint32_t lo = base[kk*ldt + n];
    uint32_t hi = base[(kk+1)*ldt + n];
    f.u[v] = lo | (hi << 16);
  }
  return f.v;
}
// A fragment from A^T stored row-major (K x M): base -> A^T[k0][m0].
__device__ __forceinline__ v16bf load_afrag_at(const uint16_t* base, int ldt, int lane){
  int m = lane & 15, hf = lane >> 4;
  BF16Frag f;
#pragma unroll
  for (int v = 0; v < 8; ++v){
    int kb = (v < 4 ? 2*v : 16 + 2*(v - 4)) + 8*hf;
    uint32_t lo = base[kb*ldt + m];
    uint32_t hi = base[(kb+1)*ldt + m];
    f.u[v] = lo | (hi << 16);
  }
  return f.v;
}

__device__ __forceinline__ float wave_sum(float s){
#pragma unroll
  for (int off = 16; off > 0; off >>= 1) s += __shfl_xor(s, off, 32);
  return s;
}

// ============================================================================
// f32 -> bf16 pair conversion (grid-stride over pairs).
// ============================================================================
__global__ __launch_bounds__(256) void cvt_bf16_kernel(const float* __restrict__ in,
                                                       uint16_t* __restrict__ outp,
                                                       long npairs)
{
  long i = (long)blockIdx.x*256 + threadIdx.x;
  long stride = (long)gridDim.x*256;
  for (; i < npairs; i += stride)
    *(uint32_t*)&outp[2*i] = pack_bf16(in[2*i], in[2*i+1]);
}

// ============================================================================
// Wt[n*K + kf] = bf16(W[kf*N + n]) : transpose + convert, 32x32 LDS tiles.
// ============================================================================
__global__ __launch_bounds__(256) void transpose_bf16_kernel(const float* __restrict__ W,
                                                             uint16_t* __restrict__ Wt,
                                                             int K, int N)
{
  __shared__ float tile[32][33];
  int n0 = blockIdx.x*32, k0 = blockIdx.y*32;
  int lx = threadIdx.x & 31, ly = threadIdx.x >> 5;  // 32 x 8
  for (int yy = ly; yy < 32; yy += 8)
    tile[yy][lx] = W[(size_t)(k0+yy)*N + n0 + lx];
  __syncthreads();
  for (int yy = ly; yy < 32; yy += 8)
    Wt[(size_t)(n0+yy)*K + k0 + lx] = (uint16_t)f2bf_bits(tile[lx][yy]);
}

// ============================================================================
// GEMM: Y(MxN,f32) = Xb(MxK,bf16 row-major) @ Wt^T (Wt is NxK bf16 row-major).
// 128x128 block tile, 8 waves (each 32x64), K staged 64-deep, double-buffered
// via global_load_async_to_lds_b128 (ASYNCcnt), overlapping DMA with WMMA.
// NOTE: stage buffers are pure offsets from the single extern-shared base so
// InferAddressSpaces keeps the fragment loads on the DS (LDS) path.
// ============================================================================
#define GTM 128
#define GTN 128
#define GKS 64     // K-depth per stage
#define GLDA 72    // bf16 element stride in LDS (144B rows: 16B-aligned + padded)

__global__ __launch_bounds__(256) void gemm_bb(const uint16_t* __restrict__ Xb,
                                               const uint16_t* __restrict__ Wt,
                                               float* __restrict__ Y,
                                               int M, int N, int K)
{
  extern __shared__ char gsm[];
  uint16_t* lds = (uint16_t*)gsm;
  const uint32_t dynBase = __builtin_amdgcn_groupstaticsize();
  const uint32_t tileBytes = 128*GLDA*2;

  const int tid = threadIdx.x, lane = tid & 31, wave = tid >> 5;
  const int m0 = blockIdx.y * GTM, n0 = blockIdx.x * GTN;
  const int wr = wave >> 1, wc = wave & 1;

  v8f acc[2][4] = {};

  // This thread's 4 chunks of the 128x64 bf16 tile (16B per chunk).
  int cr[4], ccb[4];
#pragma unroll
  for (int j = 0; j < 4; ++j){ int idx = tid + 256*j; cr[j] = idx >> 3; ccb[j] = (idx & 7) << 3; }

  const uint16_t* Agp = Xb + (size_t)m0*K;
  const uint16_t* Bgp = Wt + (size_t)n0*K;

  auto issue_stage = [&](int ks, int buf){
    const uint16_t* ga = Agp + (size_t)ks*GKS;
    const uint16_t* gb = Bgp + (size_t)ks*GKS;
    const uint32_t aoff = dynBase + (uint32_t)buf*tileBytes;
    const uint32_t boff = dynBase + (uint32_t)(2 + buf)*tileBytes;
#pragma unroll
    for (int j = 0; j < 4; ++j)
      async_ldst_b128(aoff + (uint32_t)(cr[j]*GLDA + ccb[j])*2,
                      (uint64_t)(uintptr_t)(ga + (size_t)cr[j]*K + ccb[j]));
#pragma unroll
    for (int j = 0; j < 4; ++j)
      async_ldst_b128(boff + (uint32_t)(cr[j]*GLDA + ccb[j])*2,
                      (uint64_t)(uintptr_t)(gb + (size_t)cr[j]*K + ccb[j]));
  };

  issue_stage(0, 0);
  wait_async0();
  __syncthreads();

  const int nk = K / GKS;
  for (int ks = 0; ks < nk; ++ks){
    const int cur = ks & 1;
    if (ks + 1 < nk) issue_stage(ks + 1, cur ^ 1);
    // Offset arithmetic on the shared base (addrspace-inference friendly).
    const uint16_t* Ac = lds + (size_t)cur*(128*GLDA);
    const uint16_t* Bc = lds + (size_t)(2 + cur)*(128*GLDA);
#pragma unroll
    for (int kk = 0; kk < GKS; kk += 32){
      v16bf afr[2], bfr[4];
#pragma unroll
      for (int rt = 0; rt < 2; ++rt) afr[rt] = load_afrag(Ac + (wr*32 + rt*16)*GLDA + kk, GLDA, lane);
#pragma unroll
      for (int ct = 0; ct < 4; ++ct) bfr[ct] = load_bfrag_bt(Bc + (wc*64 + ct*16)*GLDA + kk, GLDA, lane);
#pragma unroll
      for (int rt = 0; rt < 2; ++rt)
#pragma unroll
        for (int ct = 0; ct < 4; ++ct)
          acc[rt][ct] = wmma_bf16(afr[rt], bfr[ct], acc[rt][ct]);
    }
    wait_async0();     // next-stage DMA (this wave's share) landed
    __syncthreads();   // everyone done reading cur + writing next
  }

  // D layout: row-in-tile = v + 8*(lane>>4), col-in-tile = lane&15
  const int hf = lane >> 4;
  const int cN = lane & 15;
#pragma unroll
  for (int rt = 0; rt < 2; ++rt)
#pragma unroll
    for (int ct = 0; ct < 4; ++ct)
#pragma unroll
      for (int v = 0; v < 8; ++v){
        int row = m0 + wr*32 + rt*16 + v + 8*hf;
        int col = n0 + wc*64 + ct*16 + cN;
        Y[(size_t)row*N + col] = acc[rt][ct][v];
      }
}

// ============================================================================
// Gate: ld[token,h] = log_sigmoid(x[token,:] @ Wf[:,h] + delta[h])
// ============================================================================
__global__ __launch_bounds__(256) void fgate_kernel(const float* __restrict__ x,
                                                    const float* __restrict__ Wf,
                                                    const float* __restrict__ delta,
                                                    float* __restrict__ ld)
{
  __shared__ float xs[DD];
  const int tid = threadIdx.x;
  const int token = blockIdx.x;
  const float* xr = x + (size_t)token*DD;
  for (int i = tid; i < DD; i += 256) xs[i] = xr[i];
  __syncthreads();
  const int lane = tid & 31, head = tid >> 5;
  float s = 0.f;
  for (int i = lane; i < DD; i += 32) s += xs[i] * Wf[i*HH + head];
  s = wave_sum(s);
  if (lane == 0){
    float z = s + delta[head];
    ld[(size_t)token*HH + head] = fminf(z, 0.f) - log1pf(expf(-fabsf(z)));
  }
}

// ============================================================================
// In-place silu + L2 normalize per 128-element row. One wave per row.
// ============================================================================
__global__ __launch_bounds__(256) void silu_l2norm_kernel(float* __restrict__ p, int nrows)
{
  int row = blockIdx.x * 8 + (threadIdx.x >> 5);
  if (row >= nrows) return;
  int lane = threadIdx.x & 31;
  float* r = p + (size_t)row * HDIM;
  float vals[4]; float ss = 0.f;
#pragma unroll
  for (int j = 0; j < 4; ++j){
    float xv = r[lane*4 + j];
    float sv = xv / (1.f + expf(-xv));
    vals[j] = sv; ss += sv*sv;
  }
  ss = wave_sum(ss);
  float inv = 1.f / fmaxf(sqrtf(ss), 1e-12f);
#pragma unroll
  for (int j = 0; j < 4; ++j) r[lane*4 + j] = vals[j]*inv;
}

// ============================================================================
// Per-head RMSNorm; reads f32, writes bf16 rows (feeds the final GEMM).
// ============================================================================
__global__ __launch_bounds__(256) void rmsnorm_bf16_kernel(const float* __restrict__ p,
                                                           const float* __restrict__ w,
                                                           uint16_t* __restrict__ ob,
                                                           int nrows)
{
  int row = blockIdx.x * 8 + (threadIdx.x >> 5);
  if (row >= nrows) return;
  int lane = threadIdx.x & 31;
  int head = row % HH;
  const float* r = p + (size_t)row * HDIM;
  const float* wr = w + head*HDIM;
  float vals[4]; float ss = 0.f;
#pragma unroll
  for (int j = 0; j < 4; ++j){ vals[j] = r[lane*4 + j]; ss += vals[j]*vals[j]; }
  ss = wave_sum(ss);
  float inv = 1.f / sqrtf(ss/(float)HDIM + 1e-6f);
  float o0 = vals[0]*inv*wr[lane*4+0];
  float o1 = vals[1]*inv*wr[lane*4+1];
  float o2 = vals[2]*inv*wr[lane*4+2];
  float o3 = vals[3]*inv*wr[lane*4+3];
  uint16_t* orow = ob + (size_t)row*HDIM;
  *(uint32_t*)&orow[lane*4 + 0] = pack_bf16(o0, o1);
  *(uint32_t*)&orow[lane*4 + 2] = pack_bf16(o2, o3);
}

// ============================================================================
// Chunked scan. One block per (b,h). S (128x128 f32) lives in LDS for the
// whole sequence. Per chunk: scores + decay mask, u = v + e^cl * q@S,
// forward substitution of (I-A)out=u, then S = e^cl_end * S + (k.w)^T @ out.
// ============================================================================
#define ALS 132   // f32 stride for S
#define ALB 132   // stride for 64-row f32/bf16 chunk tiles
#define ALA 68    // stride for 64x64 A matrix

__global__ __launch_bounds__(256) void attn_scan_kernel(const float* __restrict__ q,
                                                        const float* __restrict__ k,
                                                        const float* __restrict__ vv,
                                                        const float* __restrict__ ld,
                                                        float* __restrict__ ao)
{
  extern __shared__ char smem_raw[];
  float* S    = (float*)smem_raw;        // 128 x ALS f32  (recurrent state)
  float* ub   = S + 128*ALS;             // 64 x ALB f32   (u, then solved out)
  float* Amat = ub + 64*ALB;             // 64 x ALA f32   (decayed scores)
  float* cl   = Amat + 64*ALA;           // 64             (cumulative log-decay)
  float* ldv  = cl + 64;                 // 64
  uint16_t* Sb  = (uint16_t*)(ldv + 64); // 128 x ALB bf16 (S snapshot)
  uint16_t* qb  = Sb  + 128*ALB;         // 64 x ALB bf16
  uint16_t* kb  = qb  + 64*ALB;          // 64 x ALB bf16
  uint16_t* kwb = kb  + 64*ALB;          // 64 x ALB bf16  (k scaled by w_end)
  uint16_t* ob  = kwb + 64*ALB;          // 64 x ALB bf16  (out snapshot)

  const int tid  = threadIdx.x;
  const int lane = tid & 31, wave = tid >> 5;
  const int hf = lane >> 4, m = lane & 15;
  const int bh = blockIdx.x;
  const int b = bh / HH, h = bh % HH;
  const size_t tokBase = (size_t)b * NN;

  for (int i = tid; i < 128*ALS; i += 256) S[i] = 0.f;
  __syncthreads();

  for (int ci = 0; ci < NN/CK; ++ci){
    const int t0 = ci * CK;

    // ---- Phase 0: stage q/k (bf16), u=v (f32), gates; snapshot S -> bf16 ----
    for (int i = tid; i < CK*(HDIM/2); i += 256){
      int r = i >> 6; int c = (i & 63) << 1;
      size_t gi = (tokBase + t0 + r)*DD + h*HDIM + c;
      *(uint32_t*)&qb[r*ALB + c] = pack_bf16(q[gi], q[gi+1]);
      *(uint32_t*)&kb[r*ALB + c] = pack_bf16(k[gi], k[gi+1]);
      ub[r*ALB + c    ] = vv[gi];
      ub[r*ALB + c + 1] = vv[gi+1];
    }
    for (int i = tid; i < 128*(HDIM/2); i += 256){
      int r = i >> 6; int c = (i & 63) << 1;
      *(uint32_t*)&Sb[r*ALB + c] = pack_bf16(S[r*ALS + c], S[r*ALS + c + 1]);
    }
    if (tid < CK) ldv[tid] = ld[(tokBase + t0 + tid)*HH + h];
    __syncthreads();
    if (tid == 0){
      float run = 0.f;
      for (int i = 0; i < CK; ++i){ run += ldv[i]; cl[i] = run; }
    }
    __syncthreads();

    // ---- Phase 2a: scores = q @ k^T, apply strict-causal decay -> Amat ----
#pragma unroll
    for (int t = 0; t < 2; ++t){
      int idx = wave*2 + t;
      int ti = idx >> 2, tj = idx & 3;
      v8f acc = {};
#pragma unroll
      for (int kk = 0; kk < HDIM; kk += 32){
        v16bf a  = load_afrag   (qb + (ti*16)*ALB + kk, ALB, lane);
        v16bf bb = load_bfrag_bt(kb + (tj*16)*ALB + kk, ALB, lane);
        acc = wmma_bf16(a, bb, acc);
      }
#pragma unroll
      for (int v = 0; v < 8; ++v){
        int i = ti*16 + v + 8*hf;
        int j = tj*16 + m;
        Amat[i*ALA + j] = (j < i) ? acc[v] * expf(cl[i] - cl[j]) : 0.f;
      }
    }
    // ---- Phase 2b: u += exp(cl_i) * (q @ S) ----
#pragma unroll
    for (int t = 0; t < 4; ++t){
      int idx = wave*4 + t;
      int ti = idx >> 3, tj = idx & 7;
      v8f acc = {};
#pragma unroll
      for (int kk = 0; kk < HDIM; kk += 32){
        v16bf a  = load_afrag  (qb + (ti*16)*ALB + kk, ALB, lane);
        v16bf bb = load_bfrag_b(Sb + kk*ALB + tj*16, ALB, lane);
        acc = wmma_bf16(a, bb, acc);
      }
#pragma unroll
      for (int v = 0; v < 8; ++v){
        int i = ti*16 + v + 8*hf;
        int f = tj*16 + m;
        ub[i*ALB + f] += expf(cl[i]) * acc[v];
      }
    }
    __syncthreads();

    // ---- Phase 3: forward substitution (I - A) out = u; thread owns col f ----
    if (tid < HDIM){
      int f = tid;
      for (int i = 1; i < CK; ++i){
        float s = ub[i*ALB + f];
        for (int j = 0; j < i; ++j) s += Amat[i*ALA + j] * ub[j*ALB + f];
        ub[i*ALB + f] = s;   // only this thread ever touches column f
      }
    }
    __syncthreads();

    // ---- Phase 4: scale S by e^cl_end; build kw (bf16) and out (bf16) ----
    {
      float clend = cl[CK-1];
      float g = expf(clend);
      for (int i = tid; i < 128*HDIM; i += 256){
        int r = i >> 7, c = i & 127;
        S[r*ALS + c] *= g;
      }
      for (int i = tid; i < CK*(HDIM/2); i += 256){
        int r = i >> 6, c = (i & 63) << 1;
        float w = expf(clend - cl[r]);
        float k0 = bf16_to_f32(kb[r*ALB + c    ]) * w;
        float k1 = bf16_to_f32(kb[r*ALB + c + 1]) * w;
        *(uint32_t*)&kwb[r*ALB + c] = pack_bf16(k0, k1);
        *(uint32_t*)&ob [r*ALB + c] = pack_bf16(ub[r*ALB + c], ub[r*ALB + c + 1]);
      }
    }
    __syncthreads();

    // ---- Phase 5: S += kw^T @ out  (C/D read/write the f32 state tiles) ----
    {
      int ti = wave;  // e-rows 16*wave..16*wave+15
#pragma unroll
      for (int tj = 0; tj < 8; ++tj){
        v8f acc;
#pragma unroll
        for (int v = 0; v < 8; ++v) acc[v] = S[(ti*16 + v + 8*hf)*ALS + tj*16 + m];
#pragma unroll
        for (int kk = 0; kk < CK; kk += 32){
          v16bf a  = load_afrag_at(kwb + kk*ALB + ti*16, ALB, lane);
          v16bf bb = load_bfrag_b (ob  + kk*ALB + tj*16, ALB, lane);
          acc = wmma_bf16(a, bb, acc);
        }
#pragma unroll
        for (int v = 0; v < 8; ++v) S[(ti*16 + v + 8*hf)*ALS + tj*16 + m] = acc[v];
      }
    }
    // write chunk output
    for (int i = tid; i < CK*HDIM; i += 256){
      int r = i >> 7, c = i & 127;
      ao[(tokBase + t0 + r)*DD + h*HDIM + c] = ub[r*ALB + c];
    }
    __syncthreads();
  }
}

// ============================================================================
extern "C" void kernel_launch(void* const* d_in, const int* in_sizes, int n_in,
                              void* d_out, int out_size, void* d_ws, size_t ws_size,
                              hipStream_t stream)
{
  (void)in_sizes; (void)n_in; (void)out_size; (void)ws_size;
  const float* x     = (const float*)d_in[0];
  const float* Wq    = (const float*)d_in[1];
  const float* Wk    = (const float*)d_in[2];
  const float* Wv    = (const float*)d_in[3];
  const float* Wf    = (const float*)d_in[4];
  const float* Wo    = (const float*)d_in[5];
  const float* delta = (const float*)d_in[6];
  const float* nw    = (const float*)d_in[7];
  float* out = (float*)d_out;

  char* ws = (char*)d_ws;
  size_t off = 0;
  auto wsalloc = [&](size_t bytes) -> void* {
    void* p = (void*)(ws + off);
    off += (bytes + 255) & ~(size_t)255;
    return p;
  };
  float*    q   = (float*)   wsalloc(sizeof(float)*(size_t)NTOK*DD);
  float*    kk  = (float*)   wsalloc(sizeof(float)*(size_t)NTOK*DD);
  float*    v   = (float*)   wsalloc(sizeof(float)*(size_t)NTOK*DD);
  float*    ao  = (float*)   wsalloc(sizeof(float)*(size_t)NTOK*DD);
  float*    ld  = (float*)   wsalloc(sizeof(float)*(size_t)NTOK*HH);
  uint16_t* xb  = (uint16_t*)wsalloc(sizeof(uint16_t)*(size_t)NTOK*DD);
  uint16_t* aob = (uint16_t*)wsalloc(sizeof(uint16_t)*(size_t)NTOK*DD);
  uint16_t* wtq = (uint16_t*)wsalloc(sizeof(uint16_t)*(size_t)DD*DD);
  uint16_t* wtk = (uint16_t*)wsalloc(sizeof(uint16_t)*(size_t)DD*DD);
  uint16_t* wtv = (uint16_t*)wsalloc(sizeof(uint16_t)*(size_t)DD*DD);
  uint16_t* wto = (uint16_t*)wsalloc(sizeof(uint16_t)*(size_t)DD*DD);

  // bf16 operand preparation (memory-bound, one pass)
  cvt_bf16_kernel<<<2048, 256, 0, stream>>>(x, xb, (long)NTOK*DD/2);
  dim3 tg(DD/32, DD/32);
  transpose_bf16_kernel<<<tg, 256, 0, stream>>>(Wq, wtq, DD, DD);
  transpose_bf16_kernel<<<tg, 256, 0, stream>>>(Wk, wtk, DD, DD);
  transpose_bf16_kernel<<<tg, 256, 0, stream>>>(Wv, wtv, DD, DD);
  transpose_bf16_kernel<<<tg, 256, 0, stream>>>(Wo, wto, DD, DD);

  dim3 gg(DD/GTN, NTOK/GTM);
  size_t gsmem = (size_t)4*128*GLDA*2;   // 72KB double-buffered stages
  gemm_bb<<<gg, 256, gsmem, stream>>>(xb, wtq, q,  NTOK, DD, DD);
  gemm_bb<<<gg, 256, gsmem, stream>>>(xb, wtk, kk, NTOK, DD, DD);
  gemm_bb<<<gg, 256, gsmem, stream>>>(xb, wtv, v,  NTOK, DD, DD);

  fgate_kernel<<<NTOK, 256, 0, stream>>>(x, Wf, delta, ld);
  silu_l2norm_kernel<<<(NTOK*HH)/8, 256, 0, stream>>>(q,  NTOK*HH);
  silu_l2norm_kernel<<<(NTOK*HH)/8, 256, 0, stream>>>(kk, NTOK*HH);

  size_t smem = (size_t)(128*ALS + 64*ALB + 64*ALA + 128)*sizeof(float)
              + (size_t)(128*ALB + 4*64*ALB)*sizeof(uint16_t);
  attn_scan_kernel<<<BB*HH, 256, smem, stream>>>(q, kk, v, ld, ao);

  rmsnorm_bf16_kernel<<<(NTOK*HH)/8, 256, 0, stream>>>(ao, nw, aob, NTOK*HH);
  gemm_bb<<<gg, 256, gsmem, stream>>>(aob, wto, out, NTOK, DD, DD);
}
